// SSMCore_13915694039734
// MI455X (gfx1250) — matrix operand order repaired and checked by reading, not compiled, not run on hardware
//
#include <hip/hip_runtime.h>
#include <hip/hip_bf16.h>
#include <stdint.h>

// Problem constants (from reference)
#define BATCH   2
#define TLEN    2048
#define DMODEL  1024
#define DINNER  2048
#define NSTATE  16
#define DCONV   4
#define DTRANK  64
#define PROJN   96           // DTRANK + 2*NSTATE
#define ROWS    (BATCH*TLEN) // 4096

#define NSEG    32
#define SEGT    (TLEN/NSEG)  // 64

typedef __attribute__((ext_vector_type(16))) __bf16 v16bf;
typedef __attribute__((ext_vector_type(8)))  __bf16 v8bf;
typedef __attribute__((ext_vector_type(4)))  __bf16 v4bf;
typedef __attribute__((ext_vector_type(8)))  float  v8f;
typedef __attribute__((ext_vector_type(4)))  float  f4;

static __device__ __forceinline__ __bf16 to_bf16(float f) {
  unsigned u = __builtin_bit_cast(unsigned, f);
  u += 0x7FFFu + ((u >> 16) & 1u);                 // round-to-nearest-even
  unsigned short h = (unsigned short)(u >> 16);
  return __builtin_bit_cast(__bf16, h);
}
static __device__ __forceinline__ float bf16_to_f32(__bf16 h) {
  unsigned u = ((unsigned)__builtin_bit_cast(unsigned short, h)) << 16;
  return __builtin_bit_cast(float, u);
}

// ---- CDNA5 async global->LDS copy (ASYNCcnt-tracked), 16B per lane -------
// Generic pointers to LDS carry the LDS byte offset in their low 32 bits
// (flat aperture rule).
static __device__ __forceinline__ void async_b128(unsigned lds_addr, const void* g) {
  unsigned long long ga = (unsigned long long)(size_t)g;
  asm volatile("global_load_async_to_lds_b128 %0, %1, off"
               :: "v"(lds_addr), "v"(ga) : "memory");
}
static __device__ __forceinline__ void wait_async0() {
  asm volatile("s_wait_asynccnt 0x0" ::: "memory");
}
static __device__ __forceinline__ void wait_async4() {
  asm volatile("s_wait_asynccnt 0x4" ::: "memory");
}

// ---------------------------------------------------------------------------
// Tiled GEMM: C[M,N] = A[M,K] * Bt[N,K]^T, bf16 in, fp32 accum (WMMA).
// A: bf16 row-major (lda).  Bt: bf16, B pre-transposed to N x K (ldt).
// 256 threads = 8 wave32; tile 128x128x32; wave computes 64x32 = 4x2 WMMA.
// Double-buffered LDS filled by global_load_async_to_lds_b128; steady-state
// loop is branch-free (last tile peeled for the ASYNCcnt drain).
// Requires M%128==0, K%32==0; N ragged (address-clamped loads, guarded store).
// ---------------------------------------------------------------------------
#define BM 128
#define BN 128
#define BK 32

__global__ void __launch_bounds__(256)
gemm_bf16_wmma(const __bf16* __restrict__ A, const __bf16* __restrict__ Bt,
               void* __restrict__ Cout, int M, int N, int K,
               int lda, int ldt, int ldc, int storeBf16)
{
  __shared__ alignas(32) __bf16 As[2][BM][BK];
  __shared__ alignas(32) __bf16 Bs[2][BN][BK];

  const int tid  = threadIdx.x;
  const int lane = tid & 31;
  const int wave = tid >> 5;
  const int m0   = blockIdx.y * BM;
  const int n0   = blockIdx.x * BN;
  const int wm   = (wave >> 2) * 64;
  const int wn   = (wave & 3) * 32;
  const int l15  = lane & 15;
  const int hi   = lane >> 4;

  v8f acc[4][2];
#pragma unroll
  for (int i = 0; i < 4; ++i)
#pragma unroll
    for (int j = 0; j < 2; ++j)
      acc[i][j] = (v8f){0.f,0.f,0.f,0.f,0.f,0.f,0.f,0.f};

  // issue one 128x32 A tile + 128x32 Bt tile: 4 async b128 per thread-position
  auto issue_tile = [&](int kt, int buf) {
    const int k0 = kt * BK;
#pragma unroll
    for (int p = 0; p < 2; ++p) {
      int cidx = tid + p * 256;          // 0..511 16B-chunks
      int r   = cidx >> 2;               // 0..127
      int c8  = (cidx & 3) * 8;          // 0,8,16,24 (halves)
      async_b128((unsigned)(size_t)(void*)&As[buf][r][c8],
                 A + (size_t)(m0 + r) * lda + k0 + c8);
    }
#pragma unroll
    for (int p = 0; p < 2; ++p) {
      int cidx = tid + p * 256;
      int r   = cidx >> 2;
      int c8  = (cidx & 3) * 8;
      int rn  = n0 + r;                  // clamp ragged N: no divergence,
      if (rn > N - 1) rn = N - 1;        // dup rows land in unstored cols
      async_b128((unsigned)(size_t)(void*)&Bs[buf][r][c8],
                 Bt + (size_t)rn * ldt + k0 + c8);
    }
  };

  // 4x2 WMMA on one resident tile pair
  auto compute_tile = [&](int buf) {
    v16bf bfrag[2];
#pragma unroll
    for (int j = 0; j < 2; ++j)
      bfrag[j] = *(const v16bf*)&Bs[buf][wn + j * 16 + l15][hi * 16];
#pragma unroll
    for (int ii = 0; ii < 4; ++ii) {
      union { v16bf v; v8bf h[2]; } a;
      const __bf16* arow = &As[buf][wm + ii * 16 + l15][0];
      a.h[0] = *(const v8bf*)(arow + hi * 16);       // K = hi16 + 0..7
      a.h[1] = *(const v8bf*)(arow + hi * 16 + 8);   // K = hi16 + 8..15
#pragma unroll
      for (int j = 0; j < 2; ++j)
        acc[ii][j] = __builtin_amdgcn_wmma_f32_16x16x32_bf16(
            false, a.v, false, bfrag[j], (short)0, acc[ii][j], false, false);
    }
  };

  const int ntiles = K / BK;
  issue_tile(0, 0);

  // steady state: branch-free software pipeline
  for (int i = 0; i < ntiles - 1; ++i) {
    const int buf = i & 1;
    issue_tile(i + 1, buf ^ 1);
    wait_async4();                 // previous batch (this tile) has landed
    __syncthreads();
    compute_tile(buf);
    __syncthreads();
  }
  // drain: last tile
  wait_async0();
  __syncthreads();
  compute_tile((ntiles - 1) & 1);

  // store: lane l15 = col, VGPR v -> row = v + hi*8
#pragma unroll
  for (int i = 0; i < 4; ++i) {
#pragma unroll
    for (int j = 0; j < 2; ++j) {
      int col = n0 + wn + j * 16 + l15;
      if (col < N) {
#pragma unroll
        for (int v = 0; v < 8; ++v) {
          int row = m0 + wm + i * 16 + hi * 8 + v;
          if (storeBf16)
            ((__bf16*)Cout)[(size_t)row * ldc + col] = to_bf16(acc[i][j][v]);
          else
            ((float*)Cout)[(size_t)row * ldc + col] = acc[i][j][v];
        }
      }
    }
  }
}

// ---------------------------------------------------------------------------
// fp32 -> bf16 streaming convert (n % 4 == 0)
// ---------------------------------------------------------------------------
__global__ void __launch_bounds__(256)
cvt_kernel(const float* __restrict__ src, __bf16* __restrict__ dst, int n)
{
  int i = (blockIdx.x * 256 + threadIdx.x) * 4;
  if (i < n) {
    f4 v = *(const f4*)(src + i);
    v4bf h;
    h[0] = to_bf16(v[0]); h[1] = to_bf16(v[1]);
    h[2] = to_bf16(v[2]); h[3] = to_bf16(v[3]);
    *(v4bf*)(dst + i) = h;
  }
}

// fp32 W[K,N] -> bf16 Wt[N,K] (LDS-tiled transpose, 32x32)
__global__ void __launch_bounds__(256)
cvt_transpose_kernel(const float* __restrict__ W, __bf16* __restrict__ Wt,
                     int K, int N)
{
  __shared__ float tile[32][33];
  const int tid = threadIdx.x;
  const int kb = blockIdx.y * 32, nb = blockIdx.x * 32;
#pragma unroll
  for (int p = 0; p < 4; ++p) {
    int r = (tid >> 5) + p * 8;          // k
    int c = tid & 31;                    // n
    float v = 0.f;
    if (kb + r < K && nb + c < N) v = W[(size_t)(kb + r) * N + nb + c];
    tile[r][c] = v;
  }
  __syncthreads();
#pragma unroll
  for (int p = 0; p < 4; ++p) {
    int r = (tid >> 5) + p * 8;          // n
    int c = tid & 31;                    // k
    if (nb + r < N && kb + c < K)
      Wt[(size_t)(nb + r) * K + kb + c] = to_bf16(tile[c][r]);
  }
}

// proj[:, 0:64] fp32 -> dtbf [ROWS,64] bf16
__global__ void __launch_bounds__(256)
cvt_dt_kernel(const float* __restrict__ proj, __bf16* __restrict__ dtbf)
{
  int idx = blockIdx.x * 256 + threadIdx.x;   // ROWS*64
  int r = idx >> 6, c = idx & 63;
  dtbf[idx] = to_bf16(proj[(size_t)r * PROJN + c]);
}

// ---------------------------------------------------------------------------
// Causal depthwise conv1d (K=4) + SiLU; u = xzbf[:, :DINNER] (bf16).
// Writes uc fp32 (scan) and ucbf (next GEMM).
// ---------------------------------------------------------------------------
__global__ void __launch_bounds__(256)
conv_silu_kernel(const __bf16* __restrict__ xzbf, const float* __restrict__ cw,
                 const float* __restrict__ cb, float* __restrict__ ucf,
                 __bf16* __restrict__ ucbf)
{
  int idx = blockIdx.x * 256 + threadIdx.x;   // over ROWS*DINNER
  int d   = idx & (DINNER - 1);
  int bt  = idx >> 11;
  int t   = bt & (TLEN - 1);
  float acc = cb[d];
#pragma unroll
  for (int k = 0; k < DCONV; ++k) {
    int tt = t + k - (DCONV - 1);
    if (tt >= 0)
      acc += bf16_to_f32(xzbf[(size_t)(bt - t + tt) * (2 * DINNER) + d]) *
             cw[d * DCONV + k];
  }
  float s = acc / (1.f + __expf(-acc));       // SiLU
  ucf[(size_t)bt * DINNER + d]  = s;
  ucbf[(size_t)bt * DINNER + d] = to_bf16(s);
}

// ---------------------------------------------------------------------------
// Chunked selective scan.  mode 0: per-segment (P = prod dA, q = h | h0=0).
// mode 1: re-run with correct h0, emit y, fuse D-skip + SiLU(z) gate -> bf16.
// ---------------------------------------------------------------------------
__global__ void __launch_bounds__(256)
seg_scan_kernel(const float* __restrict__ dlin, const float* __restrict__ bdt,
                const float* __restrict__ A_log, const float* __restrict__ proj,
                const float* __restrict__ ucf, const float* __restrict__ Dvec,
                const __bf16* __restrict__ xzbf,
                float* __restrict__ Pbuf, float* __restrict__ qbuf,
                const float* __restrict__ h0buf, __bf16* __restrict__ ygbf,
                int mode)
{
  const int s    = blockIdx.x & (NSEG - 1);
  const int rest = blockIdx.x >> 5;
  const int b    = rest >> 3;
  const int d    = (rest & 7) * 256 + threadIdx.x;
  const int bd   = b * DINNER + d;

  __shared__ float sBC[SEGT][2 * NSTATE];
#pragma unroll
  for (int i = 0; i < (SEGT * 2 * NSTATE) / 256; ++i) {
    int idx = threadIdx.x + i * 256;
    int tr = idx >> 5, c = idx & 31;
    sBC[tr][c] = proj[(size_t)(b * TLEN + s * SEGT + tr) * PROJN + DTRANK + c];
  }
  __syncthreads();

  float Ac[NSTATE];
#pragma unroll
  for (int n = 0; n < NSTATE; ++n) Ac[n] = -__expf(A_log[d * NSTATE + n]);
  const float bdd = bdt[d];
  const float Dd  = Dvec[d];

  float h[NSTATE], p[NSTATE];
#pragma unroll
  for (int n = 0; n < NSTATE; ++n) { h[n] = 0.f; p[n] = 1.f; }
  if (mode) {
#pragma unroll
    for (int n = 0; n < NSTATE; ++n)
      h[n] = h0buf[((size_t)s * ROWS + bd) * NSTATE + n];
  }

  for (int i = 0; i < SEGT; ++i) {
    size_t row = (size_t)b * TLEN + s * SEGT + i;
    float xv    = dlin[row * DINNER + d] + bdd;
    float delta = xv > 20.f ? xv : log1pf(__expf(xv));  // softplus
    float u     = ucf[row * DINNER + d];
    float du    = delta * u;
    float y     = 0.f;
#pragma unroll
    for (int n = 0; n < NSTATE; ++n) {
      float da = __expf(delta * Ac[n]);
      h[n] = da * h[n] + du * sBC[i][n];
      if (mode) y += h[n] * sBC[i][NSTATE + n];
      else      p[n] *= da;
    }
    if (mode) {
      y += u * Dd;
      float z = bf16_to_f32(xzbf[row * (size_t)(2 * DINNER) + DINNER + d]);
      ygbf[row * DINNER + d] = to_bf16(y * (z / (1.f + __expf(-z))));
    }
  }
  if (!mode) {
#pragma unroll
    for (int n = 0; n < NSTATE; ++n) {
      Pbuf[((size_t)s * ROWS + bd) * NSTATE + n] = p[n];
      qbuf[((size_t)s * ROWS + bd) * NSTATE + n] = h[n];
    }
  }
}

__global__ void __launch_bounds__(256)
seg_combine_kernel(const float* __restrict__ Pbuf, const float* __restrict__ qbuf,
                   float* __restrict__ h0buf)
{
  int bd = blockIdx.x * 256 + threadIdx.x;   // 0..4095
  float H[NSTATE];
#pragma unroll
  for (int n = 0; n < NSTATE; ++n) H[n] = 0.f;
  for (int sIt = 0; sIt < NSEG; ++sIt) {
    size_t base = ((size_t)sIt * ROWS + bd) * NSTATE;
#pragma unroll
    for (int n = 0; n < NSTATE; ++n) {
      h0buf[base + n] = H[n];
      H[n] = Pbuf[base + n] * H[n] + qbuf[base + n];
    }
  }
}

// ---------------------------------------------------------------------------
extern "C" void kernel_launch(void* const* d_in, const int* in_sizes, int n_in,
                              void* d_out, int out_size, void* d_ws, size_t ws_size,
                              hipStream_t stream) {
  const float* x      = (const float*)d_in[0];
  const float* W_in   = (const float*)d_in[1];
  const float* conv_w = (const float*)d_in[2];
  const float* conv_b = (const float*)d_in[3];
  const float* W_xprj = (const float*)d_in[4];
  const float* W_dt   = (const float*)d_in[5];
  const float* b_dt   = (const float*)d_in[6];
  const float* A_log  = (const float*)d_in[7];
  const float* Dv     = (const float*)d_in[8];
  const float* W_out  = (const float*)d_in[9];
  float* out = (float*)d_out;

  // workspace carve-out (256B-aligned blocks)
  char* wsp = (char*)d_ws;
  size_t off = 0;
  auto alloc = [&](size_t bytes) {
    void* p = wsp + off;
    off = (off + bytes + 255) & ~(size_t)255;
    return p;
  };
  __bf16* xzbf = (__bf16*)alloc((size_t)ROWS * 2 * DINNER * 2);  // 32 MB
  float*  ucf  = (float*) alloc((size_t)ROWS * DINNER * 4);      // 32 MB
  __bf16* ucbf = (__bf16*)alloc((size_t)ROWS * DINNER * 2);      // 16 MB
  float*  proj = (float*) alloc((size_t)ROWS * PROJN * 4);
  float*  dlin = (float*) alloc((size_t)ROWS * DINNER * 4);      // 32 MB
  float*  Pb   = (float*) alloc((size_t)NSEG * ROWS * NSTATE * 4);
  float*  qb   = (float*) alloc((size_t)NSEG * ROWS * NSTATE * 4);
  float*  h0b  = (float*) alloc((size_t)NSEG * ROWS * NSTATE * 4);
  __bf16* xbf  = (__bf16*)alloc((size_t)ROWS * DMODEL * 2);
  __bf16* Wtin = (__bf16*)alloc((size_t)2 * DINNER * DMODEL * 2);
  __bf16* Wtxp = (__bf16*)alloc((size_t)PROJN * DINNER * 2);
  __bf16* Wtdt = (__bf16*)alloc((size_t)DINNER * DTRANK * 2);
  __bf16* Wtou = (__bf16*)alloc((size_t)DMODEL * DINNER * 2);
  __bf16* dtbf = (__bf16*)alloc((size_t)ROWS * DTRANK * 2);
  __bf16* ygbf = (__bf16*)alloc((size_t)ROWS * DINNER * 2);      // 16 MB

  dim3 blk(256);

  // 0) one-time bf16 conversions (+ weight transposes to Bt[N][K])
  cvt_kernel<<<dim3((ROWS * DMODEL / 4 + 255) / 256), blk, 0, stream>>>(
      x, xbf, ROWS * DMODEL);
  cvt_transpose_kernel<<<dim3(2 * DINNER / 32, DMODEL / 32), blk, 0, stream>>>(
      W_in, Wtin, DMODEL, 2 * DINNER);
  cvt_transpose_kernel<<<dim3(PROJN / 32, DINNER / 32), blk, 0, stream>>>(
      W_xprj, Wtxp, DINNER, PROJN);
  cvt_transpose_kernel<<<dim3(DINNER / 32, DTRANK / 32), blk, 0, stream>>>(
      W_dt, Wtdt, DTRANK, DINNER);
  cvt_transpose_kernel<<<dim3(DMODEL / 32, DINNER / 32), blk, 0, stream>>>(
      W_out, Wtou, DINNER, DMODEL);

  // 1) xz = x @ W_in           (4096 x 4096 x 1024) -> bf16
  gemm_bf16_wmma<<<dim3(2 * DINNER / BN, ROWS / BM), blk, 0, stream>>>(
      xbf, Wtin, xzbf, ROWS, 2 * DINNER, DMODEL, DMODEL, DMODEL, 2 * DINNER, 1);

  // 2) uc = silu(conv1d(u))
  conv_silu_kernel<<<dim3((ROWS * DINNER) / 256), blk, 0, stream>>>(
      xzbf, conv_w, conv_b, ucf, ucbf);

  // 3) proj = uc @ W_xproj     (4096 x 96 x 2048) -> fp32
  gemm_bf16_wmma<<<dim3(1, ROWS / BM), blk, 0, stream>>>(
      ucbf, Wtxp, proj, ROWS, PROJN, DINNER, DINNER, DINNER, PROJN, 0);

  // 4) dlin = dt @ W_dt        (4096 x 2048 x 64) -> fp32
  cvt_dt_kernel<<<dim3((ROWS * DTRANK) / 256), blk, 0, stream>>>(proj, dtbf);
  gemm_bf16_wmma<<<dim3(DINNER / BN, ROWS / BM), blk, 0, stream>>>(
      dtbf, Wtdt, dlin, ROWS, DINNER, DTRANK, DTRANK, DTRANK, DINNER, 0);

  // 5) chunked scan: P,q -> h0 -> y + gate (bf16)
  seg_scan_kernel<<<dim3(BATCH * (DINNER / 256) * NSEG), blk, 0, stream>>>(
      dlin, b_dt, A_log, proj, ucf, Dv, xzbf, Pb, qb, h0b, ygbf, 0);
  seg_combine_kernel<<<dim3(ROWS / 256), blk, 0, stream>>>(Pb, qb, h0b);
  seg_scan_kernel<<<dim3(BATCH * (DINNER / 256) * NSEG), blk, 0, stream>>>(
      dlin, b_dt, A_log, proj, ucf, Dv, xzbf, Pb, qb, h0b, ygbf, 1);

  // 6) out = y_gated @ W_out   (4096 x 1024 x 2048) -> fp32
  gemm_bf16_wmma<<<dim3(DMODEL / BN, ROWS / BM), blk, 0, stream>>>(
      ygbf, Wtou, out, ROWS, DMODEL, DINNER, DINNER, DINNER, DMODEL, 0);
}
